// MLA_84069689852142
// MI455X (gfx1250) — matrix operand order repaired and checked by reading, not compiled, not run on hardware
//
#include <hip/hip_runtime.h>
#include <hip/hip_bf16.h>

// ---------------- problem constants ----------------
#define S_LEN   2048
#define D_MODEL 2048
#define KVRANK  512
#define QRANK   1536
#define NH      16
#define D_R     64
#define D_N     128
#define D_V     128
#define D_QK    192   // D_R + D_N

typedef __attribute__((ext_vector_type(16))) __bf16        v16bf;
typedef __attribute__((ext_vector_type(8)))  float         v8f;
typedef __attribute__((ext_vector_type(4)))  unsigned int  u32x4;

union Frag { u32x4 u[2]; v16bf v; };

__device__ __forceinline__ unsigned short f2bf(float f) {
  unsigned int u = __builtin_bit_cast(unsigned int, f);
  u += 0x7fffu + ((u >> 16) & 1u);          // round-to-nearest-even
  return (unsigned short)(u >> 16);
}
__device__ __forceinline__ unsigned int pk2(float a, float b) {
  return (unsigned int)f2bf(a) | ((unsigned int)f2bf(b) << 16);
}
__device__ __forceinline__ v8f wmma_bf16(v16bf a, v16bf b, v8f c) {
  return __builtin_amdgcn_wmma_f32_16x16x32_bf16(
      /*neg_a=*/false, a, /*neg_b=*/false, b,
      /*c_mod=*/(short)0, c, /*reuse_a=*/false, /*reuse_b=*/false);
}

// =====================================================================
// Tiled GEMM: out[M,N] = A[M,K] * B[K,N], bf16 WMMA compute, f32 accum.
//   A_BF16    : A is bf16 row-major (else fp32 row-major, converted on stage)
//   OUT_TBF16 : store bf16 transposed out[N,M] (for V^T), else fp32 row-major
//   BN        : block tile N extent (128 normal, 64 for the D_R projection)
// Block: 256 threads = 8 waves; block tile 128 x BN; wave tile 32 x (BN/2).
// Intensity at BN=128: 2*128*128*K flops / (256*K*4B) = 32 flop/byte.
// =====================================================================
template<bool A_BF16, bool OUT_TBF16, int BN>
__global__ __launch_bounds__(256) void gemm_k(
    const void* __restrict__ Ap, const float* __restrict__ B,
    void* __restrict__ Out, int M, int N, int K)
{
  constexpr int JT = BN / 32;            // 16-wide j tiles per wave (2 or 4)
  constexpr int NV = BN / 4;             // float4 per B row
  __shared__ __align__(16) unsigned short sA[128 * 40];  // [row][k], pad 40
  __shared__ __align__(16) unsigned short sB[BN * 40];   // [n][k] (B transposed)
  const int tid  = threadIdx.x;
  const int wave = tid >> 5, lane = tid & 31;
  const int l16 = lane & 15, lh = lane >> 4;
  const int m0 = blockIdx.y * 128, n0 = blockIdx.x * BN;
  const int wm = (wave >> 1) * 32, wn = (wave & 1) * (BN / 2);
  v8f acc[2][JT] = {};

  for (int k0 = 0; k0 < K; k0 += 32) {
    __syncthreads();
    // ---- stage A tile 128x32 ----
    if (A_BF16) {
      const unsigned short* A = (const unsigned short*)Ap;
#pragma unroll
      for (int it = 0; it < 2; ++it) {
        int lin = tid + it * 256;          // 512 vectors of 8 bf16
        int row = lin >> 2, cv = lin & 3;
        u32x4 d = *(const u32x4*)(A + (size_t)(m0 + row) * K + k0 + cv * 8);
        *(u32x4*)(&sA[row * 40 + cv * 8]) = d;
        if (k0 + 32 < K)
          __builtin_prefetch(A + (size_t)(m0 + row) * K + k0 + 32 + cv * 8, 0, 0);
      }
    } else {
      const float* A = (const float*)Ap;
#pragma unroll
      for (int it = 0; it < 4; ++it) {
        int lin = tid + it * 256;          // 1024 float4
        int row = lin >> 3, cv = lin & 7;
        float4 d = *(const float4*)(A + (size_t)(m0 + row) * K + k0 + cv * 4);
        *(unsigned int*)(&sA[row * 40 + cv * 4])     = pk2(d.x, d.y);
        *(unsigned int*)(&sA[row * 40 + cv * 4 + 2]) = pk2(d.z, d.w);
        if (k0 + 32 < K)
          __builtin_prefetch(A + (size_t)(m0 + row) * K + k0 + 32 + cv * 4, 0, 0);
      }
    }
    // ---- stage B tile 32xBN transposed -> sB[n][k] ----
#pragma unroll
    for (int it = 0; it < JT; ++it) {
      int lin = tid + it * 256;            // 8*BN float4
      int kk = lin / NV, nv = lin % NV;
      float4 d = *(const float4*)(B + (size_t)(k0 + kk) * N + n0 + nv * 4);
      sB[(nv * 4 + 0) * 40 + kk] = f2bf(d.x);
      sB[(nv * 4 + 1) * 40 + kk] = f2bf(d.y);
      sB[(nv * 4 + 2) * 40 + kk] = f2bf(d.z);
      sB[(nv * 4 + 3) * 40 + kk] = f2bf(d.w);
      if (k0 + 32 < K)
        __builtin_prefetch(B + (size_t)(k0 + 32 + kk) * N + n0 + nv * 4, 0, 0);
    }
    __syncthreads();
    // ---- fragments (wave32; lane half h holds K 16h..16h+15 for both A and B) ----
    Frag af[2], bfr[JT];
#pragma unroll
    for (int i = 0; i < 2; ++i) {          // A: lane -> row l16; 16 contiguous K per half
      int row = wm + i * 16 + l16, ko = lh * 16;
      af[i].u[0] = *(const u32x4*)(&sA[row * 40 + ko]);
      af[i].u[1] = *(const u32x4*)(&sA[row * 40 + ko + 8]);
    }
#pragma unroll
    for (int j = 0; j < JT; ++j) {         // B: lane -> col l16; 16 contiguous K per half
      int n = wn + j * 16 + l16, ko = lh * 16;
      bfr[j].u[0] = *(const u32x4*)(&sB[n * 40 + ko]);
      bfr[j].u[1] = *(const u32x4*)(&sB[n * 40 + ko + 8]);
    }
#pragma unroll
    for (int i = 0; i < 2; ++i)
#pragma unroll
      for (int j = 0; j < JT; ++j)
        acc[i][j] = wmma_bf16(af[i].v, bfr[j].v, acc[i][j]);
  }
  // ---- epilogue: C layout lane -> (M = lh*8+r, N = l16) ----
#pragma unroll
  for (int i = 0; i < 2; ++i)
#pragma unroll
    for (int j = 0; j < JT; ++j)
#pragma unroll
      for (int r = 0; r < 8; ++r) {
        int Mr = m0 + wm + i * 16 + lh * 8 + r;
        int Nc = n0 + wn + j * 16 + l16;
        if (OUT_TBF16)
          ((unsigned short*)Out)[(size_t)Nc * M + Mr] = f2bf(acc[i][j][r]);
        else
          ((float*)Out)[(size_t)Mr * N + Nc] = acc[i][j][r];
      }
}

// =====================================================================
// RoPE + concat packers -> bf16 [S][H][192]
// =====================================================================
__global__ __launch_bounds__(256) void pack_k_kernel(
    const float* __restrict__ krr,   // [S][64]
    const float* __restrict__ kn,    // [S][H*128]
    unsigned short* __restrict__ kcat)
{
  int s = blockIdx.x, t = threadIdx.x;
  __shared__ unsigned short ksm[64];
  if (t < 32) {
    float inv_freq = __expf(-(float)(2 * t) / 64.f * 9.2103403719762f); // theta^{-2i/64}
    float sn, cs;
    __sincosf((float)s * inv_freq, &sn, &cs);
    float te = krr[(size_t)s * 64 + 2 * t];
    float to = krr[(size_t)s * 64 + 2 * t + 1];
    ksm[2 * t]     = f2bf(te * cs - to * sn);
    ksm[2 * t + 1] = f2bf(te * sn + to * cs);
  }
  __syncthreads();
  for (int i = t; i < NH * D_QK; i += 256) {
    int hh = i / D_QK, d = i % D_QK;
    unsigned short v = (d < D_R) ? ksm[d]
                                 : f2bf(kn[(size_t)s * (NH * D_N) + hh * D_N + (d - D_R)]);
    kcat[(size_t)s * (NH * D_QK) + i] = v;
  }
}

__global__ __launch_bounds__(256) void pack_q_kernel(
    const float* __restrict__ qrr,   // [S][H*64]
    const float* __restrict__ qn,    // [S][H*128]
    unsigned short* __restrict__ qcat)
{
  int s = blockIdx.x, t = threadIdx.x;
  __shared__ unsigned short qsm[NH * D_R];
  const float scale = 0.072168783648703f;    // 1/sqrt(192), folded into q
  for (int i = t; i < NH * 32; i += 256) {   // 512 rope pairs
    int hh = i >> 5, p = i & 31;
    float inv_freq = __expf(-(float)(2 * p) / 64.f * 9.2103403719762f);
    float sn, cs;
    __sincosf((float)s * inv_freq, &sn, &cs);
    float te = qrr[(size_t)s * (NH * D_R) + hh * D_R + 2 * p];
    float to = qrr[(size_t)s * (NH * D_R) + hh * D_R + 2 * p + 1];
    qsm[hh * D_R + 2 * p]     = f2bf((te * cs - to * sn) * scale);
    qsm[hh * D_R + 2 * p + 1] = f2bf((te * sn + to * cs) * scale);
  }
  __syncthreads();
  for (int i = t; i < NH * D_QK; i += 256) {
    int hh = i / D_QK, d = i % D_QK;
    unsigned short v = (d < D_R) ? qsm[hh * D_R + d]
                                 : f2bf(qn[(size_t)s * (NH * D_N) + hh * D_N + (d - D_R)] * scale);
    qcat[(size_t)s * (NH * D_QK) + i] = v;
  }
}

// =====================================================================
// Flash attention: grid (S/64, H); 4 waves, each owns 16 q rows x full DV.
// K/V^T fragments streamed from global (L2-resident: 192MB L2 >> 21MB K+V).
// Online softmax; P transposed C->A layout through per-wave LDS.
// =====================================================================
__global__ __launch_bounds__(128) void attn_kernel(
    const unsigned short* __restrict__ qcat,  // [S][H][192] bf16 (pre-scaled)
    const unsigned short* __restrict__ kcat,  // [S][H][192] bf16
    const unsigned short* __restrict__ vT,    // [H*128][S]  bf16
    unsigned short* __restrict__ aout)        // [S][H*128]  bf16
{
  __shared__ __align__(16) unsigned short sP[4][16 * 40];
  const int h    = blockIdx.y;
  const int wave = threadIdx.x >> 5, lane = threadIdx.x & 31;
  const int l16 = lane & 15, lh = lane >> 4;
  const int qbase = blockIdx.x * 64 + wave * 16;
  unsigned short* myP = sP[wave];

  // Q fragments for this wave's 16 rows (6 K-chunks of 32)
  Frag qf[6];
  const unsigned short* qrow = qcat + ((size_t)(qbase + l16) * NH + h) * D_QK;
#pragma unroll
  for (int c = 0; c < 6; ++c) {
    qf[c].u[0] = *(const u32x4*)(qrow + c * 32 + lh * 16);
    qf[c].u[1] = *(const u32x4*)(qrow + c * 32 + lh * 16 + 8);
  }

  v8f oacc[8] = {};
  float mrow[8], lrow[8];
#pragma unroll
  for (int r = 0; r < 8; ++r) { mrow[r] = -1e30f; lrow[r] = 0.f; }

  for (int kb = 0; kb < S_LEN; kb += 32) {
    // ---- scores: two 16x16 tiles over 32 keys ----
    v8f s0 = {}, s1 = {};
    const unsigned short* k0p = kcat + ((size_t)(kb + l16) * NH + h) * D_QK + lh * 16;
    const unsigned short* k1p = kcat + ((size_t)(kb + 16 + l16) * NH + h) * D_QK + lh * 16;
#pragma unroll
    for (int c = 0; c < 6; ++c) {
      Frag b0, b1;
      b0.u[0] = *(const u32x4*)(k0p + c * 32);
      b0.u[1] = *(const u32x4*)(k0p + c * 32 + 8);
      b1.u[0] = *(const u32x4*)(k1p + c * 32);
      b1.u[1] = *(const u32x4*)(k1p + c * 32 + 8);
      s0 = wmma_bf16(qf[c].v, b0.v, s0);
      s1 = wmma_bf16(qf[c].v, b1.v, s1);
    }
    // ---- online softmax (row = lh*8+r, spread over 16 lanes sharing lh) ----
    float alpha[8];
#pragma unroll
    for (int r = 0; r < 8; ++r) {
      float mx = fmaxf(s0[r], s1[r]);
      mx = fmaxf(mx, __shfl_xor(mx, 1));
      mx = fmaxf(mx, __shfl_xor(mx, 2));
      mx = fmaxf(mx, __shfl_xor(mx, 4));
      mx = fmaxf(mx, __shfl_xor(mx, 8));
      float mn = fmaxf(mrow[r], mx);
      alpha[r] = __expf(mrow[r] - mn);
      mrow[r]  = mn;
      float p0 = __expf(s0[r] - mn);
      float p1 = __expf(s1[r] - mn);
      float sum = p0 + p1;
      sum += __shfl_xor(sum, 1);
      sum += __shfl_xor(sum, 2);
      sum += __shfl_xor(sum, 4);
      sum += __shfl_xor(sum, 8);
      lrow[r] = lrow[r] * alpha[r] + sum;
      myP[(lh * 8 + r) * 40 + l16]      = f2bf(p0);   // P[M][N] bf16, 16x32
      myP[(lh * 8 + r) * 40 + 16 + l16] = f2bf(p1);
    }
#pragma unroll
    for (int n = 0; n < 8; ++n)
#pragma unroll
      for (int r = 0; r < 8; ++r)
        oacc[n][r] *= alpha[r];
    asm volatile("s_wait_dscnt 0" ::: "memory");       // P visible wave-wide
    // ---- P as A fragment ----
    Frag pf;
    pf.u[0] = *(const u32x4*)(&myP[l16 * 40 + lh * 16]);
    pf.u[1] = *(const u32x4*)(&myP[l16 * 40 + lh * 16 + 8]);
    // ---- O += P @ V (V^T rows = dv, contiguous keys) ----
#pragma unroll
    for (int n = 0; n < 8; ++n) {
      const unsigned short* vp =
          vT + (size_t)(h * D_V + n * 16 + l16) * S_LEN + kb + lh * 16;
      Frag bv;
      bv.u[0] = *(const u32x4*)vp;
      bv.u[1] = *(const u32x4*)(vp + 8);
      oacc[n] = wmma_bf16(pf.v, bv.v, oacc[n]);
    }
  }
  // ---- normalize + store bf16 ----
  float inv[8];
#pragma unroll
  for (int r = 0; r < 8; ++r) inv[r] = 1.0f / lrow[r];
#pragma unroll
  for (int n = 0; n < 8; ++n)
#pragma unroll
    for (int r = 0; r < 8; ++r) {
      int row = qbase + lh * 8 + r;
      aout[(size_t)row * (NH * D_V) + h * D_V + n * 16 + l16] =
          f2bf(oacc[n][r] * inv[r]);
    }
}

// =====================================================================
// launch
// =====================================================================
extern "C" void kernel_launch(void* const* d_in, const int* in_sizes, int n_in,
                              void* d_out, int out_size, void* d_ws, size_t ws_size,
                              hipStream_t stream)
{
  (void)in_sizes; (void)n_in; (void)out_size; (void)ws_size;
  const float* x     = (const float*)d_in[0];
  const float* w_dkv = (const float*)d_in[1];
  const float* w_kr  = (const float*)d_in[2];
  const float* w_uk  = (const float*)d_in[3];
  const float* w_uv  = (const float*)d_in[4];
  const float* w_dq  = (const float*)d_in[5];
  const float* w_qr  = (const float*)d_in[6];
  const float* w_uq  = (const float*)d_in[7];
  const float* w_o   = (const float*)d_in[8];

  char* ws = (char*)d_ws;
  size_t off = 0;
  auto alloc = [&](size_t bytes) {
    void* p = ws + off;
    off += (bytes + 255) & ~(size_t)255;
    return p;
  };
  float*          c_kv   = (float*)alloc((size_t)S_LEN * KVRANK * 4);
  float*          c_q    = (float*)alloc((size_t)S_LEN * QRANK * 4);
  float*          kr_raw = (float*)alloc((size_t)S_LEN * D_R * 4);
  float*          qr_raw = (float*)alloc((size_t)S_LEN * NH * D_R * 4);
  float*          k_n    = (float*)alloc((size_t)S_LEN * NH * D_N * 4);
  float*          q_n    = (float*)alloc((size_t)S_LEN * NH * D_N * 4);
  unsigned short* vT     = (unsigned short*)alloc((size_t)NH * D_V * S_LEN * 2);
  unsigned short* qcat   = (unsigned short*)alloc((size_t)S_LEN * NH * D_QK * 2);
  unsigned short* kcat   = (unsigned short*)alloc((size_t)S_LEN * NH * D_QK * 2);
  unsigned short* aout   = (unsigned short*)alloc((size_t)S_LEN * NH * D_V * 2);

  dim3 blk(256);
  // projections (block tile 128 x 128, except the N=64 k_r projection)
  gemm_k<false, false, 128><<<dim3(KVRANK / 128,     S_LEN / 128), blk, 0, stream>>>(x,    w_dkv, c_kv,   S_LEN, KVRANK,   D_MODEL);
  gemm_k<false, false, 128><<<dim3(QRANK / 128,      S_LEN / 128), blk, 0, stream>>>(x,    w_dq,  c_q,    S_LEN, QRANK,    D_MODEL);
  gemm_k<false, false,  64><<<dim3(D_R / 64,         S_LEN / 128), blk, 0, stream>>>(x,    w_kr,  kr_raw, S_LEN, D_R,      D_MODEL);
  gemm_k<false, false, 128><<<dim3((NH * D_R) / 128, S_LEN / 128), blk, 0, stream>>>(c_q,  w_qr,  qr_raw, S_LEN, NH * D_R, QRANK);
  gemm_k<false, false, 128><<<dim3((NH * D_N) / 128, S_LEN / 128), blk, 0, stream>>>(c_q,  w_uq,  q_n,    S_LEN, NH * D_N, QRANK);
  gemm_k<false, false, 128><<<dim3((NH * D_N) / 128, S_LEN / 128), blk, 0, stream>>>(c_kv, w_uk,  k_n,    S_LEN, NH * D_N, KVRANK);
  gemm_k<false, true,  128><<<dim3((NH * D_V) / 128, S_LEN / 128), blk, 0, stream>>>(c_kv, w_uv,  vT,     S_LEN, NH * D_V, KVRANK);
  // rope + concat packing
  pack_k_kernel<<<S_LEN, 256, 0, stream>>>(kr_raw, k_n, kcat);
  pack_q_kernel<<<S_LEN, 256, 0, stream>>>(qr_raw, q_n, qcat);
  // attention
  attn_kernel<<<dim3(S_LEN / 64, NH), dim3(128), 0, stream>>>(qcat, kcat, vT, aout);
  // output projection (bf16 A path)
  gemm_k<true, false, 128><<<dim3(D_MODEL / 128, S_LEN / 128), blk, 0, stream>>>(aout, w_o, (float*)d_out, S_LEN, D_MODEL, NH * D_V);
}
